// ShapeNetModel_3882650436828
// MI455X (gfx1250) — matrix operand order repaired and checked by reading, not compiled, not run on hardware
//
#include <hip/hip_runtime.h>
#include <math.h>

// ---------------------------------------------------------------------------
// MI455X (gfx1250) DGCNN/ShapeNet. All contractions run through a WMMA GEMM
// (v_wmma_f32_16x16x32_f16, f32 accumulate): 4-wave workgroups compute a
// 64x64 output tile; the shared 32Kx64 X tile is staged through LDS
// (ds_store_b128 / ds_load_b128) so each wave's B fragments come from LDS
// while A fragments stream from global as b128 loads. Fused
// scale/residual/BN/LeakyReLU epilogues. All tile dims are exact multiples
// (ec1's K=6 is zero-padded to 32 in ws) -> no bounds checks anywhere hot.
// Workspace requirement: ~532 MiB (aliased layout below).
// ---------------------------------------------------------------------------

typedef __attribute__((ext_vector_type(16))) _Float16 v16h;
typedef __attribute__((ext_vector_type(8)))  float    v8f;

#define EP_LRELU 1
#define EP_BN    2
#define EP_RESID 4

#define LS_PAD 36   // LDS row stride (dwords) for the 64x32 tile, bank-spread

// ---- A fragment loaders -----------------------------------------------------
// A: 16 rows x 32 K. lane r=lane&15 holds row r; h=lane>>4.
//    element j: K = h*8 + j (j<8) | 16 + h*8 + (j-8) (j>=8)

// AMODE 0: row-major W (unit K stride): two contiguous 8-float runs -> 4x b128.
__device__ __forceinline__ v16h load_a_rm(const float* __restrict__ base,
                                          long ld, int row0, int k0) {
  int lane = threadIdx.x & 31;
  int r = lane & 15, h = lane >> 4;
  const float* p = base + (long)(row0 + r) * ld + k0 + h * 8;
  float4 f0 = *reinterpret_cast<const float4*>(p);
  float4 f1 = *reinterpret_cast<const float4*>(p + 4);
  float4 f2 = *reinterpret_cast<const float4*>(p + 16);
  float4 f3 = *reinterpret_cast<const float4*>(p + 20);
  v16h a;
  a[0]=(_Float16)f0.x; a[1]=(_Float16)f0.y; a[2]=(_Float16)f0.z; a[3]=(_Float16)f0.w;
  a[4]=(_Float16)f1.x; a[5]=(_Float16)f1.y; a[6]=(_Float16)f1.z; a[7]=(_Float16)f1.w;
  a[8]=(_Float16)f2.x; a[9]=(_Float16)f2.y; a[10]=(_Float16)f2.z; a[11]=(_Float16)f2.w;
  a[12]=(_Float16)f3.x; a[13]=(_Float16)f3.y; a[14]=(_Float16)f3.z; a[15]=(_Float16)f3.w;
  return a;
}

// AMODE 1: transposed W (scores: element (o,k) at o + k*ld). Lane-coalesced gather.
__device__ __forceinline__ v16h load_a_tr(const float* __restrict__ base,
                                          long ld, int row0, int k0) {
  int lane = threadIdx.x & 31;
  int r = lane & 15, h = lane >> 4;
  const float* p = base + (row0 + r) + (long)(k0 + h * 8) * ld;
  v16h a;
#pragma unroll
  for (int j = 0; j < 8; ++j)  a[j] = (_Float16)p[(long)j * ld];
  const float* p2 = p + 16 * ld;
#pragma unroll
  for (int j = 0; j < 8; ++j)  a[8 + j] = (_Float16)p2[(long)j * ld];
  return a;
}

// Generic GEMM: Y[b,o,n] = epi( scl * sum_k W(o,k) * X(k,n) ).
// 4-wave block computes 64 rows x 64 cols; wave w owns rows rowBlk+16w.
// The 32Kx64col X tile is staged in LDS as ls[col][k] (stride LS_PAD) and all
// four waves read their B fragments from it via ds_load_b128.
// BMODE 0: X row-major (elem (k,n) at k*xLd + n). BMODE 1: X^T (at n*xLd + k).
template <int AMODE, int BMODE>
__global__ void gemm_wmma_kernel(
    const float* __restrict__ W, long wLd, long wBat,
    const float* __restrict__ X, long xLd, long xBat,
    float* __restrict__ Y, long yLd, long yBat,
    const float* __restrict__ R, long rLd, long rBat,
    const float* __restrict__ bn, int bnC,
    int M, int Nc, int Kd, int flags, float scl)
{
  __shared__ float ls[64 * LS_PAD];

  int b      = blockIdx.z;
  int col0   = blockIdx.x * 64;
  int rowBlk = blockIdx.y * 64;
  int w      = threadIdx.x >> 5;
  int row0   = rowBlk + w * 16;
  const float* Wb = W + (long)b * wBat;
  const float* Xb = X + (long)b * xBat;

  int scol = threadIdx.x & 63;        // staging: column within tile
  int skq  = threadIdx.x >> 6;        // staging: k-quad phase (0..1)

  v8f acc[4] = {v8f{}, v8f{}, v8f{}, v8f{}};

  for (int k0 = 0; k0 < Kd; k0 += 32) {
    __syncthreads();
    // ---- stage X tile (32 K x 64 cols) into ls[col][k] ----
    if (BMODE == 0) {
      __builtin_prefetch(&Xb[(long)(k0 + 32) * xLd + col0 + scol], 0, 1);
#pragma unroll
      for (int it = 0; it < 4; ++it) {
        int k = (skq + it * 2) * 4;
        const float* g = Xb + (long)(k0 + k) * xLd + col0 + scol;
        float4 f;
        f.x = g[0]; f.y = g[xLd]; f.z = g[2 * xLd]; f.w = g[3 * xLd];
        *reinterpret_cast<float4*>(&ls[scol * LS_PAD + k]) = f;
      }
    } else {
      __builtin_prefetch(&Xb[(long)(col0 + scol) * xLd + k0 + 32], 0, 1);
#pragma unroll
      for (int it = 0; it < 4; ++it) {
        int k = (skq + it * 2) * 4;
        float4 f = *reinterpret_cast<const float4*>(
            Xb + (long)(col0 + scol) * xLd + k0 + k);
        *reinterpret_cast<float4*>(&ls[scol * LS_PAD + k]) = f;
      }
    }
    __syncthreads();

    v16h a = (AMODE == 0) ? load_a_rm(Wb, wLd, row0, k0)
                          : load_a_tr(Wb, wLd, row0, k0);

    int lane = threadIdx.x & 31;
    int c = lane & 15, h = lane >> 4;
#pragma unroll
    for (int t = 0; t < 4; ++t) {
      // B fragment: lane holds col 16t+c, K runs h*8..h*8+7 and 16+h*8..16+h*8+7 in ls
      const float* p = &ls[(t * 16 + c) * LS_PAD + h * 8];
      float4 f0 = *reinterpret_cast<const float4*>(p);
      float4 f1 = *reinterpret_cast<const float4*>(p + 4);
      float4 f2 = *reinterpret_cast<const float4*>(p + 16);
      float4 f3 = *reinterpret_cast<const float4*>(p + 20);
      v16h bf;
      bf[0]=(_Float16)f0.x; bf[1]=(_Float16)f0.y; bf[2]=(_Float16)f0.z; bf[3]=(_Float16)f0.w;
      bf[4]=(_Float16)f1.x; bf[5]=(_Float16)f1.y; bf[6]=(_Float16)f1.z; bf[7]=(_Float16)f1.w;
      bf[8]=(_Float16)f2.x; bf[9]=(_Float16)f2.y; bf[10]=(_Float16)f2.z; bf[11]=(_Float16)f2.w;
      bf[12]=(_Float16)f3.x; bf[13]=(_Float16)f3.y; bf[14]=(_Float16)f3.z; bf[15]=(_Float16)f3.w;
      acc[t] = __builtin_amdgcn_wmma_f32_16x16x32_f16(
          false, a, false, bf, (short)0, acc[t], false, false);
    }
  }

  int lane = threadIdx.x & 31;
  int h = lane >> 4, cl = lane & 15;
#pragma unroll
  for (int t = 0; t < 4; ++t) {
    int colb = col0 + 16 * t + cl;
#pragma unroll
    for (int i = 0; i < 8; ++i) {
      int row = row0 + h * 8 + i;
      float v = acc[t][i] * scl;
      if (flags & EP_RESID) v += R[(long)b * rBat + (long)row * rLd + colb];
      if (flags & EP_BN) {
        float g  = bn[row];
        float bb = bn[bnC + row];
        float mm = bn[2 * bnC + row];
        float vv = bn[3 * bnC + row];
        v = (v - mm) * (g * rsqrtf(vv + 1e-5f)) + bb;
      }
      if (flags & EP_LRELU) v = (v >= 0.f) ? v : 0.2f * v;
      Y[(long)b * yBat + (long)row * yLd + colb] = v;
    }
  }
}

// Zero-pad ec1_w1 (64x6) into a 64x32 buffer so its GEMM needs no K bounds.
__global__ void pad_w1_kernel(const float* __restrict__ w, float* __restrict__ wp)
{
  int t = blockIdx.x * blockDim.x + threadIdx.x;
  if (t >= 64 * 32) return;
  int o = t >> 5, c = t & 31;
  wp[t] = (c < 6) ? w[o * 6 + c] : 0.f;
}

// kNN: per query point, full-N squared-distance sweep with LDS tiles of 128
// candidate points; top-32 via replace-worst.
template <int C>
__global__ void topk_kernel(const float* __restrict__ x, long chS, long bS,
                            int* __restrict__ idx, int N)
{
  __shared__ float sf[C][128];
  __shared__ float ssq[128];
  int b = blockIdx.y;
  int tid = threadIdx.x;
  int n = blockIdx.x * 128 + tid;
  const float* xb = x + (long)b * bS;

  float q[C];
  float sqn = 0.f;
#pragma unroll
  for (int c = 0; c < C; ++c) {
    float v = xb[(long)c * chS + n];
    q[c] = v; sqn += v * v;
  }

  float bd[32]; int bi[32];
#pragma unroll
  for (int j = 0; j < 32; ++j) { bd[j] = 3.402823466e38f; bi[j] = 0; }
  float worst = 3.402823466e38f; int wslot = 0;

  for (int m0 = 0; m0 < N; m0 += 128) {
    __syncthreads();
    {
      float s = 0.f;
      for (int c = 0; c < C; ++c) {
        float v = xb[(long)c * chS + (m0 + tid)];
        sf[c][tid] = v; s += v * v;
      }
      ssq[tid] = s;
    }
    __syncthreads();
    for (int mm = 0; mm < 128; ++mm) {
      float dot = 0.f;
#pragma unroll
      for (int c = 0; c < C; ++c) dot += q[c] * sf[c][mm];
      float d = sqn + ssq[mm] - 2.f * dot;
      if (d < worst) {
        bd[wslot] = d; bi[wslot] = m0 + mm;
        worst = bd[0]; wslot = 0;
#pragma unroll
        for (int j = 1; j < 32; ++j)
          if (bd[j] > worst) { worst = bd[j]; wslot = j; }
      }
    }
  }
  long base = ((long)b * N + n) * 32;
#pragma unroll
  for (int j = 0; j < 32; ++j) idx[base + j] = bi[j];
}

// Edge features: E[b, c, n*32+k] = x[nbr]-x[n] (c<Cin) | x[n] (c>=Cin);
// rows [2*Cin, padRows) are zeroed (ec1 K-padding).
__global__ void build_edge_kernel(const float* __restrict__ x, long chS, long bS,
                                  const int* __restrict__ idx,
                                  float* __restrict__ E, int Cin, int padRows, int N)
{
  long NK = (long)N * 32;
  long col = (long)blockIdx.x * blockDim.x + threadIdx.x;
  int b = blockIdx.y;
  if (col >= NK) return;
  int n = (int)(col >> 5);
  int m = idx[(long)b * NK + col];
  const float* xb = x + (long)b * bS;
  float* Eb = E + (long)b * padRows * NK;
  for (int c = 0; c < Cin; ++c) {
    float xn = xb[(long)c * chS + n];
    float xm = xb[(long)c * chS + m];
    Eb[(long)c * NK + col]         = xm - xn;
    Eb[(long)(c + Cin) * NK + col] = xn;
  }
  for (int c = 2 * Cin; c < padRows; ++c) Eb[(long)c * NK + col] = 0.f;
}

// Max over the 32 neighbor columns of each (channel, point).
__global__ void kmax_kernel(const float* __restrict__ H2, float* __restrict__ out,
                            long outCh, long outBat, int N)
{
  int t = blockIdx.x * blockDim.x + threadIdx.x;
  int b = blockIdx.y;
  if (t >= 64 * N) return;
  int c = t / N, n = t % N;
  const float* h = H2 + ((long)b * 64 + c) * (long)N * 32 + (long)n * 32;
  float m = h[0];
#pragma unroll
  for (int k = 1; k < 32; ++k) m = fmaxf(m, h[k]);
  out[(long)b * outBat + (long)c * outCh + n] = m;
}

// Row softmax over N entries; one 256-thread block per (b, n) row.
__global__ void softmax_kernel(float* __restrict__ S, int N)
{
  int b = blockIdx.y, n = blockIdx.x, tid = threadIdx.x;
  float* row = S + ((long)b * N + n) * (long)N;
  __shared__ float red[8];

  float mx = -3.402823466e38f;
  for (int i = tid; i < N; i += 256) mx = fmaxf(mx, row[i]);
  for (int o = 16; o > 0; o >>= 1) mx = fmaxf(mx, __shfl_xor(mx, o, 32));
  if ((tid & 31) == 0) red[tid >> 5] = mx;
  __syncthreads();
  if (tid == 0) { float m = red[0]; for (int j = 1; j < 8; ++j) m = fmaxf(m, red[j]); red[0] = m; }
  __syncthreads();
  mx = red[0];
  __syncthreads();

  float sum = 0.f;
  for (int i = tid; i < N; i += 256) { float e = __expf(row[i] - mx); row[i] = e; sum += e; }
  for (int o = 16; o > 0; o >>= 1) sum += __shfl_xor(sum, o, 32);
  if ((tid & 31) == 0) red[tid >> 5] = sum;
  __syncthreads();
  if (tid == 0) { float s = 0.f; for (int j = 0; j < 8; ++j) s += red[j]; red[0] = s; }
  __syncthreads();
  float inv = 1.f / red[0];
  for (int i = tid; i < N; i += 256) row[i] *= inv;
}

static inline void launch_gemm(hipStream_t s, int amode, int bmode,
    const float* W, long wLd, long wB,
    const float* X, long xLd, long xB,
    float* Y, long yLd, long yB,
    const float* R, long rLd, long rB,
    const float* bn, int bnC,
    int M, int Nc, int Kd, int flags, float scl, int batches)
{
  dim3 grid((unsigned)(Nc / 64), (unsigned)(M / 64), (unsigned)batches);
  if (amode == 0 && bmode == 0)
    gemm_wmma_kernel<0, 0><<<grid, 128, 0, s>>>(W, wLd, wB, X, xLd, xB, Y, yLd, yB,
                                                R, rLd, rB, bn, bnC, M, Nc, Kd, flags, scl);
  else if (amode == 1)
    gemm_wmma_kernel<1, 0><<<grid, 128, 0, s>>>(W, wLd, wB, X, xLd, xB, Y, yLd, yB,
                                                R, rLd, rB, bn, bnC, M, Nc, Kd, flags, scl);
  else
    gemm_wmma_kernel<0, 1><<<grid, 128, 0, s>>>(W, wLd, wB, X, xLd, xB, Y, yLd, yB,
                                                R, rLd, rB, bn, bnC, M, Nc, Kd, flags, scl);
}

extern "C" void kernel_launch(void* const* d_in, const int* in_sizes, int n_in,
                              void* d_out, int out_size, void* d_ws, size_t ws_size,
                              hipStream_t stream)
{
  (void)in_sizes; (void)n_in; (void)out_size; (void)ws_size;
  const int  B = 8, N = 2048;
  const long NK = (long)N * 32; // 65536

  const float* x       = (const float*)d_in[0];
  const float* ec1_w1  = (const float*)d_in[1];
  const float* ec1_bn1 = (const float*)d_in[2];
  const float* ec1_w2  = (const float*)d_in[3];
  const float* ec1_bn2 = (const float*)d_in[4];
  const float* ec2_w1  = (const float*)d_in[5];
  const float* ec2_bn1 = (const float*)d_in[6];
  const float* ec2_w2  = (const float*)d_in[7];
  const float* ec2_bn2 = (const float*)d_in[8];
  const float* ec3_w1  = (const float*)d_in[9];
  const float* ec3_bn1 = (const float*)d_in[10];
  const float* ec3_w2  = (const float*)d_in[11];
  const float* ec3_bn2 = (const float*)d_in[12];
  const float* emb_w   = (const float*)d_in[13];
  const float* q_w     = (const float*)d_in[14];
  const float* k_w     = (const float*)d_in[15];
  const float* v_w     = (const float*)d_in[16];
  const float* att_bn1 = (const float*)d_in[17];
  const float* ff_w1   = (const float*)d_in[18];
  const float* ff_w2   = (const float*)d_in[19];
  const float* att_bn2 = (const float*)d_in[20];
  const float* cb_w    = (const float*)d_in[21];
  const float* cb_bn   = (const float*)d_in[22];

  // Workspace layout (bytes). Total required: 532 MiB.
  const long MiB = 1LL << 20;
  char* wsb   = (char*)d_ws;
  int*   idx  = (int*)(wsb + 0);              //   2 MiB: B*N*32 int
  float* W1P  = (float*)(wsb + 2 * MiB);      //   8 KiB: padded ec1_w1 (64x32)
  float* XC   = (float*)(wsb + 4 * MiB);      //  12 MiB: (B,192,N) f32
  float* E    = (float*)(wsb + 20 * MiB);     // 256 MiB: edge feats / later h,q,k,v
  float* H    = (float*)(wsb + 276 * MiB);    // 128 MiB: MLP hidden / later scores
  float* H2   = (float*)(wsb + 404 * MiB);    // 128 MiB: MLP out / later hbn1,h2

  // Aliases for the attention phase (EdgeConv scratch is dead by then).
  float* hbuf  = E;                                   // (B,1024,N) 64 MiB each
  float* qbuf  = (float*)((char*)E + 64 * MiB);
  float* kbuf  = (float*)((char*)E + 128 * MiB);
  float* vbuf  = (float*)((char*)E + 192 * MiB);
  float* Sbuf  = H;                                   // (B,N,N) 128 MiB
  float* hbn1  = H2;
  float* h2    = (float*)((char*)H2 + 64 * MiB);
  float* ffmid = E;                                   // reuses h slot after bn1

  auto edgeconv = [&](const float* fin, long chS, long bS, int Cin,
                      const float* w1, long w1Ld, const float* bn1,
                      const float* w2, const float* bn2,
                      float* fout, long outCh, long outBat)
  {
    int padRows = (Cin == 3) ? 32 : 2 * Cin;   // K padded to a multiple of 32
    dim3 gt((unsigned)(N / 128), (unsigned)B);
    if (Cin == 3) topk_kernel<3><<<gt, 128, 0, stream>>>(fin, chS, bS, idx, N);
    else          topk_kernel<64><<<gt, 128, 0, stream>>>(fin, chS, bS, idx, N);

    dim3 ge((unsigned)((NK + 255) / 256), (unsigned)B);
    build_edge_kernel<<<ge, 256, 0, stream>>>(fin, chS, bS, idx, E, Cin, padRows, N);

    // H = lrelu(bn1(w1 @ E))
    launch_gemm(stream, 0, 0, w1, w1Ld, 0, E, NK, (long)padRows * NK,
                H, NK, 64 * NK, nullptr, 0, 0, bn1, 64,
                64, (int)NK, padRows, EP_BN | EP_LRELU, 1.f, B);
    // H2 = lrelu(bn2(w2 @ H))
    launch_gemm(stream, 0, 0, w2, 64, 0, H, NK, 64 * NK,
                H2, NK, 64 * NK, nullptr, 0, 0, bn2, 64,
                64, (int)NK, 64, EP_BN | EP_LRELU, 1.f, B);
    dim3 gm((unsigned)((64 * N + 255) / 256), (unsigned)B);
    kmax_kernel<<<gm, 256, 0, stream>>>(H2, fout, outCh, outBat, N);
  };

  // Pad ec1_w1 to 64x32 (zero K rows 6..31) so the GEMM runs unchecked.
  pad_w1_kernel<<<8, 256, 0, stream>>>(ec1_w1, W1P);

  // Three EdgeConv blocks writing f1/f2/f3 into channel slices of XC (B,192,N).
  edgeconv(x,             N, 3LL * N,   3,  W1P,    32,  ec1_bn1, ec1_w2, ec1_bn2, XC,             N, 192LL * N);
  edgeconv(XC,            N, 192LL * N, 64, ec2_w1, 128, ec2_bn1, ec2_w2, ec2_bn2, XC + 64LL * N,  N, 192LL * N);
  edgeconv(XC + 64LL * N, N, 192LL * N, 64, ec3_w1, 128, ec3_bn1, ec3_w2, ec3_bn2, XC + 128LL * N, N, 192LL * N);

  // h = emb_w @ xc
  launch_gemm(stream, 0, 0, emb_w, 192, 0, XC, N, 192LL * N,
              hbuf, N, 1024LL * N, nullptr, 0, 0, nullptr, 0,
              1024, N, 192, 0, 1.f, B);
  // q, k, v
  launch_gemm(stream, 0, 0, q_w, 1024, 0, hbuf, N, 1024LL * N,
              qbuf, N, 1024LL * N, nullptr, 0, 0, nullptr, 0,
              1024, N, 1024, 0, 1.f, B);
  launch_gemm(stream, 0, 0, k_w, 1024, 0, hbuf, N, 1024LL * N,
              kbuf, N, 1024LL * N, nullptr, 0, 0, nullptr, 0,
              1024, N, 1024, 0, 1.f, B);
  launch_gemm(stream, 0, 0, v_w, 1024, 0, hbuf, N, 1024LL * N,
              vbuf, N, 1024LL * N, nullptr, 0, 0, nullptr, 0,
              1024, N, 1024, 0, 1.f, B);
  // S[b,n,m] = (1/32) * sum_c q[b,c,n] k[b,c,m]   (A = q^T: AMODE 1)
  launch_gemm(stream, 1, 0, qbuf, N, 1024LL * N, kbuf, N, 1024LL * N,
              Sbuf, N, (long)N * N, nullptr, 0, 0, nullptr, 0,
              N, N, 1024, 0, 1.f / 32.f, B);
  softmax_kernel<<<dim3((unsigned)N, (unsigned)B), 256, 0, stream>>>(Sbuf, N);
  // hbn1 = bn1(h + V @ S^T)   (B = S^T: BMODE 1; residual+BN fused)
  launch_gemm(stream, 0, 1, vbuf, N, 1024LL * N, Sbuf, N, (long)N * N,
              hbn1, N, 1024LL * N, hbuf, N, 1024LL * N, att_bn1, 1024,
              1024, N, N, EP_RESID | EP_BN, 1.f, B);
  // ffmid = lrelu(ff_w1 @ hbn1)
  launch_gemm(stream, 0, 0, ff_w1, 1024, 0, hbn1, N, 1024LL * N,
              ffmid, N, 512LL * N, nullptr, 0, 0, nullptr, 0,
              512, N, 1024, EP_LRELU, 1.f, B);
  // h2 = bn2(hbn1 + ff_w2 @ ffmid)
  launch_gemm(stream, 0, 0, ff_w2, 512, 0, ffmid, N, 512LL * N,
              h2, N, 1024LL * N, hbn1, N, 1024LL * N, att_bn2, 1024,
              1024, N, 512, EP_RESID | EP_BN, 1.f, B);
  // out = lrelu(cb_bn(cb_w @ h2))
  launch_gemm(stream, 0, 0, cb_w, 1024, 0, h2, N, 1024LL * N,
              (float*)d_out, N, 1024LL * N, nullptr, 0, 0, cb_bn, 1024,
              1024, N, 1024, EP_BN | EP_LRELU, 1.f, B);
}
